// GraphNet_19739669693138
// MI455X (gfx1250) — matrix operand order repaired and checked by reading, not compiled, not run on hardware
//
#include <hip/hip_runtime.h>

// ---------------------------------------------------------------------------
// Problem constants (match reference)
// ---------------------------------------------------------------------------
#define NBATCH 4
#define NNODES 2048
#define NEDGE  32768
#define DIN    1024
#define H1C    256
#define H2C    256
#define FDC    128
#define EPSBN  1e-5f

typedef __attribute__((ext_vector_type(16))) __bf16 bf16x16;
typedef __attribute__((ext_vector_type(8)))  __bf16 bf16x8;
typedef __attribute__((ext_vector_type(8)))  float  f32x8;

#define WMMA_BF16(a, b, c) \
  __builtin_amdgcn_wmma_f32_16x16x32_bf16(false, (a), false, (b), (short)0, (c), false, false)

// ---------------------------------------------------------------------------
// WMMA fragment helpers (layouts per CDNA5 ISA 7.12.2)
// A (16x32 bf16): lane L<16 -> row=L, K 0..7 & 16..23 ; lane L>=16 -> row=L-16,
//                 K 8..15 & 24..31.
// B (32x16 bf16, from W^T stored N-major): lane L<16 -> col=L, K 0..7 & 16..23 ;
//                 lane L>=16 -> col=L-16, K 8..15 & 24..31.
// C/D (16x16 f32): lane L -> col=L&15, rows (L<16 ? 0..7 : 8..15).
// ---------------------------------------------------------------------------
__device__ __forceinline__ bf16x16 load_a_bf16(const __bf16* A, int lda) {
  const int lane = threadIdx.x & 31;
  const int row  = lane & 15;
  const int kb   = (lane >> 4) << 3;  // 0 or 8
  const __bf16* p = A + (size_t)row * lda + kb;
  bf16x8 lo = *(const bf16x8*)(p);
  bf16x8 hi = *(const bf16x8*)(p + 16);
  return __builtin_shufflevector(lo, hi, 0, 1, 2, 3, 4, 5, 6, 7,
                                 8, 9, 10, 11, 12, 13, 14, 15);
}

__device__ __forceinline__ bf16x16 load_b_bf16(const __bf16* Wt, int ldk) {
  const int lane = threadIdx.x & 31;
  const int col  = lane & 15;
  const int kb   = (lane >> 4) << 3;  // 0 or 8
  const __bf16* p = Wt + (size_t)col * ldk + kb;
  bf16x8 lo = *(const bf16x8*)(p);
  bf16x8 hi = *(const bf16x8*)(p + 16);
  return __builtin_shufflevector(lo, hi, 0, 1, 2, 3, 4, 5, 6, 7,
                                 8, 9, 10, 11, 12, 13, 14, 15);
}

__device__ __forceinline__ void atomic_max_f32(float* addr, float v) {
  if (v >= 0.0f) atomicMax((int*)addr, __float_as_int(v));
  else           atomicMin((unsigned int*)addr, __float_as_uint(v));
}

__device__ __forceinline__ bool edge_mask(float av, int mode) {
  // mode 0: attr >= 0 (forward), 1: attr < 0 (backward), 2: all
  return (mode == 2) || (mode == 0 ? (av >= 0.0f) : (av < 0.0f));
}

// ---------------------------------------------------------------------------
// Small utility kernels
// ---------------------------------------------------------------------------
__global__ void k_fill(float* __restrict__ p, float v, int n) {
  int i = blockIdx.x * blockDim.x + threadIdx.x;
  if (i < n) p[i] = v;
}

__global__ void k_cvt_bf16(const float* __restrict__ in,
                           __bf16* __restrict__ out, int n) {
  int i = blockIdx.x * blockDim.x + threadIdx.x;
  if (i < n) out[i] = (__bf16)in[i];
}

// W (K x N row-major, f32) -> Wt (N x K row-major, bf16)
__global__ void k_transpose_bf16(const float* __restrict__ W,
                                 __bf16* __restrict__ Wt, int K, int Nn) {
  int i = blockIdx.x * blockDim.x + threadIdx.x;
  if (i >= K * Nn) return;
  int k = i / Nn, n = i % Nn;
  Wt[(size_t)n * K + k] = (__bf16)W[(size_t)k * Nn + n];
}

// segment-max finalize: -inf (untouched) -> 0
__global__ void k_finalize_max(float* __restrict__ p, int n) {
  int i = blockIdx.x * blockDim.x + threadIdx.x;
  if (i >= n) return;
  float v = p[i];
  p[i] = __builtin_isfinite(v) ? v : 0.0f;
}

// ---------------------------------------------------------------------------
// GEMM: Y[M,Nn] = A_bf16[M,K] @ Wt_bf16[Nn,K]^T + bias
// block = 128 threads (4 waves); each wave owns TWO 16-col tiles (dual acc to
// break the WMMA dependency chain, A fragment shared by both tiles).
// grid = (M/16, Nn/128)
// ---------------------------------------------------------------------------
__global__ __launch_bounds__(128) void k_gemm1(
    const __bf16* __restrict__ A, const __bf16* __restrict__ Wt,
    const float* __restrict__ bias, float* __restrict__ Y,
    int M, int Nn, int K) {
  const int wave = threadIdx.x >> 5;
  const int m0 = blockIdx.x * 16;
  const int n0 = (blockIdx.y * 8 + wave * 2) * 16;
  f32x8 acc0 = {}, acc1 = {};
  for (int k0 = 0; k0 < K; k0 += 32) {
    bf16x16 a  = load_a_bf16(A + (size_t)m0 * K + k0, K);
    bf16x16 b0 = load_b_bf16(Wt + (size_t)n0 * K + k0, K);
    bf16x16 b1 = load_b_bf16(Wt + (size_t)(n0 + 16) * K + k0, K);
    acc0 = WMMA_BF16(a, b0, acc0);
    acc1 = WMMA_BF16(a, b1, acc1);
  }
  const int lane = threadIdx.x & 31;
  const int col  = lane & 15;
  const int rb   = (lane >> 4) << 3;
  const float bv0 = bias ? bias[n0 + col] : 0.0f;
  const float bv1 = bias ? bias[n0 + 16 + col] : 0.0f;
  float* C = Y + (size_t)(m0 + rb) * Nn + n0 + col;
#pragma unroll
  for (int r = 0; r < 8; ++r) {
    C[(size_t)r * Nn]      = acc0[r] + bv0;
    C[(size_t)r * Nn + 16] = acc1[r] + bv1;
  }
}

// Y[M,Nn] = A1@W1t^T + A2@W2t^T + bias (both A bf16, same K); acc flag -> +=
// Dual n-tile per wave, interleaved WMMAs.  grid = (M/16, Nn/128)
__global__ __launch_bounds__(128) void k_gemm2(
    const __bf16* __restrict__ A1, const __bf16* __restrict__ W1t,
    const __bf16* __restrict__ A2, const __bf16* __restrict__ W2t,
    const float* __restrict__ bias, float* __restrict__ Y,
    int M, int Nn, int K, int accumulate) {
  const int wave = threadIdx.x >> 5;
  const int m0 = blockIdx.x * 16;
  const int n0 = (blockIdx.y * 8 + wave * 2) * 16;
  f32x8 acc0 = {}, acc1 = {};
  for (int k0 = 0; k0 < K; k0 += 32) {
    bf16x16 a1  = load_a_bf16(A1 + (size_t)m0 * K + k0, K);
    bf16x16 b10 = load_b_bf16(W1t + (size_t)n0 * K + k0, K);
    bf16x16 b11 = load_b_bf16(W1t + (size_t)(n0 + 16) * K + k0, K);
    acc0 = WMMA_BF16(a1, b10, acc0);
    acc1 = WMMA_BF16(a1, b11, acc1);
    bf16x16 a2  = load_a_bf16(A2 + (size_t)m0 * K + k0, K);
    bf16x16 b20 = load_b_bf16(W2t + (size_t)n0 * K + k0, K);
    bf16x16 b21 = load_b_bf16(W2t + (size_t)(n0 + 16) * K + k0, K);
    acc0 = WMMA_BF16(a2, b20, acc0);
    acc1 = WMMA_BF16(a2, b21, acc1);
  }
  const int lane = threadIdx.x & 31;
  const int col  = lane & 15;
  const int rb   = (lane >> 4) << 3;
  const float bv0 = bias ? bias[n0 + col] : 0.0f;
  const float bv1 = bias ? bias[n0 + 16 + col] : 0.0f;
  float* C = Y + (size_t)(m0 + rb) * Nn + n0 + col;
#pragma unroll
  for (int r = 0; r < 8; ++r) {
    float v0 = acc0[r] + bv0;
    float v1 = acc1[r] + bv1;
    if (accumulate) {
      v0 += C[(size_t)r * Nn];
      v1 += C[(size_t)r * Nn + 16];
    }
    C[(size_t)r * Nn]      = v0;
    C[(size_t)r * Nn + 16] = v1;
  }
}

// ---------------------------------------------------------------------------
// BatchNorm over axis=1 (nodes): per (batch, feature) stats.
// One block per batch, one thread per feature (H == blockDim.x == 256).
// ---------------------------------------------------------------------------
__global__ void k_bn_stats(const float* __restrict__ X, float* __restrict__ mu,
                           float* __restrict__ rs, int Nrows, int H) {
  const int b = blockIdx.x, h = threadIdx.x;
  const float* p = X + (size_t)b * Nrows * H + h;
  float s = 0.0f, ss = 0.0f;
  for (int i = 0; i < Nrows; ++i) {
    float v = p[(size_t)i * H];
    s += v; ss += v * v;
  }
  float m = s / (float)Nrows;
  float var = ss / (float)Nrows - m * m;
  mu[b * H + h] = m;
  rs[b * H + h] = rsqrtf(var + EPSBN);
}

__global__ void k_bn_apply(const float* __restrict__ X,
                           const float* __restrict__ mu,
                           const float* __restrict__ rs,
                           const float* __restrict__ g,
                           const float* __restrict__ be,
                           float* __restrict__ outf, __bf16* __restrict__ outb,
                           int NH, int H, int total, int relu) {
  int i = blockIdx.x * blockDim.x + threadIdx.x;
  if (i >= total) return;
  int b = i / NH;
  int h = i % H;
  float v = (X[i] - mu[b * H + h]) * rs[b * H + h] * g[h] + be[h];
  if (relu) v = v > 0.0f ? v : 0.0f;
  if (outf) outf[i] = v;
  if (outb) outb[i] = (__bf16)v;
}

// ---------------------------------------------------------------------------
// EdgeConv: for a tile of 16 edges (x batch b):
//   m  = [x_dst , x_src - x_dst]            (16 x 512, bf16, LDS)
//   h1 = relu(m @ Wa + ba)                  (16 x 256, bf16, LDS)  [WMMA K=512]
//   h2 = h1 @ Wb + bb                       (16 x 256)             [WMMA K=256]
//   masked atomic segment-max scatter of h2 rows into out[b, dst, :]
// block = 256 threads (8 waves x 2 interleaved col-tiles = 256 cols)
// ---------------------------------------------------------------------------
__global__ __launch_bounds__(256) void k_edge_conv(
    const __bf16* __restrict__ xb, const int* __restrict__ ei,
    const float* __restrict__ attr, const __bf16* __restrict__ Wat,
    const float* __restrict__ ba, const __bf16* __restrict__ Wbt,
    const float* __restrict__ bb, float* __restrict__ out, int mode) {
  __shared__ __bf16 sm_m[16][2 * H1C];   // 16 KB
  __shared__ __bf16 sm_h1[16][H1C];      //  8 KB
  const int e0 = blockIdx.x * 16;
  const int b  = blockIdx.y;
  const int tid = threadIdx.x;

  // gather endpoint features -> message tile
  for (int i = tid; i < 16 * H1C; i += 256) {
    const int le = i >> 8, h = i & (H1C - 1);
    const int e  = e0 + le;
    const int dn = ei[NEDGE + e], sn = ei[e];
    const float xi = (float)xb[((size_t)b * NNODES + dn) * H1C + h];
    const float xj = (float)xb[((size_t)b * NNODES + sn) * H1C + h];
    sm_m[le][h]       = (__bf16)xi;
    sm_m[le][H1C + h] = (__bf16)(xj - xi);
  }
  __syncthreads();

  const int wave = tid >> 5;
  const int lane = tid & 31;
  const int col  = lane & 15;
  const int rb   = (lane >> 4) << 3;
  const int n0   = wave * 32;           // this wave's 32 output columns

  // layer 1: 16x256 = m(16x512) @ Wa^T, relu, keep bf16 in LDS
  {
    f32x8 acc0 = {}, acc1 = {};
#pragma unroll
    for (int k0 = 0; k0 < 2 * H1C; k0 += 32) {
      bf16x16 a  = load_a_bf16(&sm_m[0][k0], 2 * H1C);
      bf16x16 w0 = load_b_bf16(Wat + (size_t)n0 * (2 * H1C) + k0, 2 * H1C);
      bf16x16 w1 = load_b_bf16(Wat + (size_t)(n0 + 16) * (2 * H1C) + k0, 2 * H1C);
      acc0 = WMMA_BF16(a, w0, acc0);
      acc1 = WMMA_BF16(a, w1, acc1);
    }
    const float bv0 = ba[n0 + col];
    const float bv1 = ba[n0 + 16 + col];
#pragma unroll
    for (int r = 0; r < 8; ++r) {
      float v0 = acc0[r] + bv0;
      float v1 = acc1[r] + bv1;
      sm_h1[rb + r][n0 + col]      = (__bf16)(v0 > 0.0f ? v0 : 0.0f);
      sm_h1[rb + r][n0 + 16 + col] = (__bf16)(v1 > 0.0f ? v1 : 0.0f);
    }
  }
  __syncthreads();

  // layer 2: 16x256 = h1(16x256) @ Wb^T, then masked segment-max scatter
  {
    f32x8 acc0 = {}, acc1 = {};
#pragma unroll
    for (int k0 = 0; k0 < H1C; k0 += 32) {
      bf16x16 a  = load_a_bf16(&sm_h1[0][k0], H1C);
      bf16x16 w0 = load_b_bf16(Wbt + (size_t)n0 * H1C + k0, H1C);
      bf16x16 w1 = load_b_bf16(Wbt + (size_t)(n0 + 16) * H1C + k0, H1C);
      acc0 = WMMA_BF16(a, w0, acc0);
      acc1 = WMMA_BF16(a, w1, acc1);
    }
    const float bv0 = bb[n0 + col];
    const float bv1 = bb[n0 + 16 + col];
#pragma unroll
    for (int r = 0; r < 8; ++r) {
      const int e = e0 + rb + r;
      if (edge_mask(attr[e], mode)) {
        const int dn = ei[NEDGE + e];
        float* orow = out + ((size_t)b * NNODES + dn) * H1C;
        atomic_max_f32(orow + n0 + col,      acc0[r] + bv0);
        atomic_max_f32(orow + n0 + 16 + col, acc1[r] + bv1);
      }
    }
  }
}

// ---------------------------------------------------------------------------
// SAGE aggregation pieces
// ---------------------------------------------------------------------------
__global__ void k_count(const int* __restrict__ ei,
                        const float* __restrict__ attr,
                        float* __restrict__ cnt, int mode) {
  int e = blockIdx.x * blockDim.x + threadIdx.x;
  if (e >= NEDGE) return;
  if (edge_mask(attr[e], mode)) atomicAdd(&cnt[ei[NEDGE + e]], 1.0f);
}

__global__ void k_sage_agg(const float* __restrict__ X,
                           const int* __restrict__ ei,
                           const float* __restrict__ attr,
                           float* __restrict__ S, int mode) {
  const size_t total = (size_t)NEDGE * NBATCH * H1C;
  for (size_t idx = (size_t)blockIdx.x * blockDim.x + threadIdx.x; idx < total;
       idx += (size_t)gridDim.x * blockDim.x) {
    const int h = (int)(idx % H1C);
    const size_t t = idx / H1C;
    const int b = (int)(t % NBATCH);
    const int e = (int)(t / NBATCH);
    if (!edge_mask(attr[e], mode)) continue;
    const int sn = ei[e], dn = ei[NEDGE + e];
    atomicAdd(&S[((size_t)b * NNODES + dn) * H1C + h],
              X[((size_t)b * NNODES + sn) * H1C + h]);
  }
}

__global__ void k_mean(const float* __restrict__ S,
                       const float* __restrict__ cnt,
                       __bf16* __restrict__ out, int total, int H) {
  int i = blockIdx.x * blockDim.x + threadIdx.x;
  if (i >= total) return;
  const int n = (i / H) % NNODES;
  const float c = cnt[n];
  const float inv = c > 1.0f ? 1.0f / c : 1.0f;
  out[i] = (__bf16)(S[i] * inv);
}

// ---------------------------------------------------------------------------
// Host orchestration
// ---------------------------------------------------------------------------
extern "C" void kernel_launch(void* const* d_in, const int* in_sizes, int n_in,
                              void* d_out, int out_size, void* d_ws,
                              size_t ws_size, hipStream_t stream) {
  (void)in_sizes; (void)n_in; (void)out_size; (void)ws_size;

  const float* x    = (const float*)d_in[0];
  const int*   ei   = (const int*)d_in[1];
  const float* attr = (const float*)d_in[2];
  const float* W0   = (const float*)d_in[3];
  const float* b0   = (const float*)d_in[4];
  const float* Wa[3]  = {(const float*)d_in[5],  (const float*)d_in[9],  (const float*)d_in[13]};
  const float* ba[3]  = {(const float*)d_in[6],  (const float*)d_in[10], (const float*)d_in[14]};
  const float* Wb[3]  = {(const float*)d_in[7],  (const float*)d_in[11], (const float*)d_in[15]};
  const float* bb[3]  = {(const float*)d_in[8],  (const float*)d_in[12], (const float*)d_in[16]};
  const float* Wl2  = (const float*)d_in[17];
  const float* bl2  = (const float*)d_in[18];
  const float* Wr2  = (const float*)d_in[19];
  const float* Wl3[3] = {(const float*)d_in[20], (const float*)d_in[23], (const float*)d_in[26]};
  const float* bl3[3] = {(const float*)d_in[21], (const float*)d_in[24], (const float*)d_in[27]};
  const float* Wr3[3] = {(const float*)d_in[22], (const float*)d_in[25], (const float*)d_in[28]};
  const float* g0  = (const float*)d_in[29];
  const float* be0 = (const float*)d_in[30];
  const float* g1[3]  = {(const float*)d_in[31], (const float*)d_in[33], (const float*)d_in[35]};
  const float* be1[3] = {(const float*)d_in[32], (const float*)d_in[34], (const float*)d_in[36]};
  const float* g2  = (const float*)d_in[37];
  const float* be2 = (const float*)d_in[38];

  const int M = NBATCH * NNODES;          // 8192
  const int NH = NNODES * H1C;            // per-batch element count
  const int TOT = M * H1C;                // 2,097,152
  const int TOTX = M * DIN;               // 8,388,608

  // ---- workspace carve (all 256B aligned) ----
  char* wp = (char*)d_ws;
  auto carve = [&](size_t bytes) -> void* {
    void* p = (void*)wp;
    wp += (bytes + 255) & ~(size_t)255;
    return p;
  };
  __bf16* W0t  = (__bf16*)carve((size_t)H1C * DIN * 2);
  __bf16* Wat[3], *Wbt[3], *Wl3t[3], *Wr3t[3];
  for (int i = 0; i < 3; ++i) Wat[i] = (__bf16*)carve((size_t)H1C * 2 * H1C * 2);
  for (int i = 0; i < 3; ++i) Wbt[i] = (__bf16*)carve((size_t)H1C * H1C * 2);
  __bf16* Wl2t = (__bf16*)carve((size_t)H2C * H1C * 2);
  __bf16* Wr2t = (__bf16*)carve((size_t)H2C * H1C * 2);
  for (int i = 0; i < 3; ++i) Wl3t[i] = (__bf16*)carve((size_t)FDC * H2C * 2);
  for (int i = 0; i < 3; ++i) Wr3t[i] = (__bf16*)carve((size_t)FDC * H2C * 2);
  __bf16* xbf   = (__bf16*)carve((size_t)TOTX * 2);   // x in bf16
  float*  y0    = (float*)carve((size_t)TOT * 4);
  __bf16* xbnb  = (__bf16*)carve((size_t)TOT * 2);
  float*  ec    = (float*)carve((size_t)TOT * 4);
  float*  t1f   = (float*)carve((size_t)TOT * 4);
  __bf16* t1b   = (__bf16*)carve((size_t)TOT * 2);
  float*  sbuf  = (float*)carve((size_t)TOT * 4);
  float*  cnt   = (float*)carve((size_t)NNODES * 4);
  __bf16* meanb = (__bf16*)carve((size_t)TOT * 2);
  float*  y2    = (float*)carve((size_t)TOT * 4);
  float*  t2f   = (float*)carve((size_t)TOT * 4);
  __bf16* t2b   = (__bf16*)carve((size_t)TOT * 2);
  float*  mu    = (float*)carve((size_t)NBATCH * H1C * 4);
  float*  rs    = (float*)carve((size_t)NBATCH * H1C * 4);

  auto gridE = [](int n) { return dim3((unsigned)((n + 255) / 256)); };

  // ---- weight prep: transpose + bf16; x -> bf16 ----
  auto T = [&](const float* W, __bf16* Wt, int K, int Nn) {
    k_transpose_bf16<<<gridE(K * Nn), 256, 0, stream>>>(W, Wt, K, Nn);
  };
  T(W0, W0t, DIN, H1C);
  for (int i = 0; i < 3; ++i) { T(Wa[i], Wat[i], 2 * H1C, H1C); T(Wb[i], Wbt[i], H1C, H1C); }
  T(Wl2, Wl2t, H1C, H2C);  T(Wr2, Wr2t, H1C, H2C);
  for (int i = 0; i < 3; ++i) { T(Wl3[i], Wl3t[i], H2C, FDC); T(Wr3[i], Wr3t[i], H2C, FDC); }
  k_cvt_bf16<<<gridE(TOTX), 256, 0, stream>>>(x, xbf, TOTX);

  // ---- x = BN(x @ W0 + b0) -> bf16 node features ----
  k_gemm1<<<dim3(M / 16, H1C / 128), 128, 0, stream>>>(xbf, W0t, b0, y0, M, H1C, DIN);
  k_bn_stats<<<NBATCH, H1C, 0, stream>>>(y0, mu, rs, NNODES, H1C);
  k_bn_apply<<<gridE(TOT), 256, 0, stream>>>(y0, mu, rs, g0, be0, nullptr, xbnb,
                                             NH, H1C, TOT, 0);

  // ---- output accumulator ----
  k_fill<<<gridE(M * FDC), 256, 0, stream>>>((float*)d_out, 0.0f, M * FDC);

  // ---- three branches (mode: 0 = attr>=0, 1 = attr<0, 2 = all) ----
  for (int br = 0; br < 3; ++br) {
    const int mode = br;

    // EdgeConv + segment max
    k_fill<<<gridE(TOT), 256, 0, stream>>>(ec, -__builtin_huge_valf(), TOT);
    k_edge_conv<<<dim3(NEDGE / 16, NBATCH), 256, 0, stream>>>(
        xbnb, ei, attr, Wat[br], ba[br], Wbt[br], bb[br], ec, mode);
    k_finalize_max<<<gridE(TOT), 256, 0, stream>>>(ec, TOT);

    // relu(BN(.))
    k_bn_stats<<<NBATCH, H1C, 0, stream>>>(ec, mu, rs, NNODES, H1C);
    k_bn_apply<<<gridE(TOT), 256, 0, stream>>>(ec, mu, rs, g1[br], be1[br],
                                               t1f, t1b, NH, H1C, TOT, 1);
    if (br == 1) {  // reference applies BN(g12) a second time on branch 2
      k_bn_stats<<<NBATCH, H1C, 0, stream>>>(t1f, mu, rs, NNODES, H1C);
      k_bn_apply<<<gridE(TOT), 256, 0, stream>>>(t1f, mu, rs, g1[br], be1[br],
                                                 t1f, t1b, NH, H1C, TOT, 0);
    }

    // masked degree (shared by both SAGE stages of this branch)
    k_fill<<<gridE(NNODES), 256, 0, stream>>>(cnt, 0.0f, NNODES);
    k_count<<<gridE(NEDGE), 256, 0, stream>>>(ei, attr, cnt, mode);

    // SAGE -> H2, then BN(g2)
    k_fill<<<gridE(TOT), 256, 0, stream>>>(sbuf, 0.0f, TOT);
    k_sage_agg<<<4096, 256, 0, stream>>>(t1f, ei, attr, sbuf, mode);
    k_mean<<<gridE(TOT), 256, 0, stream>>>(sbuf, cnt, meanb, TOT, H1C);
    k_gemm2<<<dim3(M / 16, H2C / 128), 128, 0, stream>>>(
        meanb, Wl2t, t1b, Wr2t, bl2, y2, M, H2C, H1C, 0);
    k_bn_stats<<<NBATCH, H2C, 0, stream>>>(y2, mu, rs, NNODES, H2C);
    k_bn_apply<<<gridE(TOT), 256, 0, stream>>>(y2, mu, rs, g2, be2,
                                               t2f, t2b, NH, H2C, TOT, 0);

    // SAGE -> FD, accumulate into d_out
    k_fill<<<gridE(TOT), 256, 0, stream>>>(sbuf, 0.0f, TOT);
    k_sage_agg<<<4096, 256, 0, stream>>>(t2f, ei, attr, sbuf, mode);
    k_mean<<<gridE(TOT), 256, 0, stream>>>(sbuf, cnt, meanb, TOT, H2C);
    k_gemm2<<<dim3(M / 16, FDC / 128), 128, 0, stream>>>(
        meanb, Wl3t[br], t2b, Wr3t[br], bl3[br], (float*)d_out, M, FDC, H2C, 1);
  }
}
